// mylstm_39805756899730
// MI455X (gfx1250) — matrix operand order repaired and checked by reading, not compiled, not run on hardware
//
#include <hip/hip_runtime.h>

// ---------------- model dims ----------------
#define HD   512
#define DIN  128
#define TSTP 64
#define WFR  240
#define NCTX 60
#define NDEC 30
#define NLAB 8

// ---------------- WMMA types ----------------
typedef __attribute__((ext_vector_type(16))) __bf16        v16bf;
typedef __attribute__((ext_vector_type(8)))  float         v8f;
typedef __attribute__((ext_vector_type(4)))  unsigned int  v4u;

union FragBF { v16bf v; v4u q[2]; unsigned short u[16]; };
union FragF  { v8f   v; float f[8]; };

__device__ __forceinline__ unsigned short f2bf(float x) {
  unsigned int u = __float_as_uint(x);
  return (unsigned short)((u + 0x7fffu + ((u >> 16) & 1u)) >> 16);
}
__device__ __forceinline__ float sigf(float x)   { return 1.0f / (1.0f + __expf(-x)); }
__device__ __forceinline__ float tanh_f(float x) { return 1.0f - 2.0f / (__expf(2.0f * x) + 1.0f); }

__device__ __forceinline__ void wmma_acc(FragF& c, const FragBF& a, const FragBF& b) {
  c.v = __builtin_amdgcn_wmma_f32_16x16x32_bf16(false, a.v, false, b.v, (short)0, c.v, false, false);
}

// B fragment (K x 16) from global bf16, row-major weight rows (lane n holds W row n0+n).
// Layout: lanes 0-15 hold K=k0..k0+7,k0+16..k0+23; lanes 16-31 hold K=k0+8..k0+15,k0+24..k0+31.
__device__ __forceinline__ FragBF load_b_bf(const unsigned short* __restrict__ W,
                                            int n0, int k0, int pitch) {
  const int lane = threadIdx.x & 31;
  const unsigned short* p = W + (size_t)(n0 + (lane & 15)) * pitch + k0 + (lane >> 4) * 8;
  FragBF f;
  f.q[0] = *(const v4u*)p;
  f.q[1] = *(const v4u*)(p + 16);
  return f;
}
// Same but converting from f32 on the fly (one-shot pre-GEMMs).
__device__ __forceinline__ FragBF load_b_f32(const float* __restrict__ W,
                                             int n0, int k0, int pitch) {
  const int lane = threadIdx.x & 31;
  const float* p = W + (size_t)(n0 + (lane & 15)) * pitch + k0 + (lane >> 4) * 8;
  FragBF f;
#pragma unroll
  for (int i = 0; i < 16; ++i) f.u[i] = f2bf(p[i + (i & 8)]);
  return f;
}
// A fragment (16 x K) from LDS row buffer (bf16, row-major, pitch in elements).
// Lane m (0-15): K chunks [k0..k0+7] and [k0+16..k0+23]; lane m+16: +8 shifted.
__device__ __forceinline__ FragBF load_a_lds(const unsigned short* p0, int k0, int pitch) {
  const int lane = threadIdx.x & 31;
  const unsigned short* p = p0 + (lane & 15) * pitch + k0 + (lane >> 4) * 8;
  FragBF f;
  f.q[0] = *(const v4u*)p;
  f.q[1] = *(const v4u*)(p + 16);
  return f;
}
// A fragment for GEMV: only matrix row 0 valid (a single LDS vector), rows 1-15 zero.
__device__ __forceinline__ FragBF load_a_vec(const unsigned short* vec, int k0) {
  const int lane = threadIdx.x & 31;
  FragBF f;
  if ((lane & 15) == 0) {
    const unsigned short* p = vec + k0 + (lane >> 4) * 8;
    f.q[0] = *(const v4u*)p;
    f.q[1] = *(const v4u*)(p + 16);
  } else {
    v4u z = {0u, 0u, 0u, 0u};
    f.q[0] = z; f.q[1] = z;
  }
  return f;
}
// A fragment from global f32 with row bound (pre-GEMMs).
__device__ __forceinline__ FragBF load_a_f32(const float* __restrict__ A,
                                             int m0, int M, int k0, int pitch) {
  const int lane = threadIdx.x & 31;
  const int row = m0 + (lane & 15);
  const int hs = lane >> 4;
  FragBF f;
  if (row < M) {
    const float* p = A + (size_t)row * pitch + k0 + hs * 8;
#pragma unroll
    for (int i = 0; i < 8; ++i) f.u[i] = f2bf(p[i]);
#pragma unroll
    for (int i = 0; i < 8; ++i) f.u[8 + i] = f2bf(p[16 + i]);
  } else {
#pragma unroll
    for (int i = 0; i < 16; ++i) f.u[i] = 0;
  }
  return f;
}

// ---------------- prep kernels ----------------
__global__ void k_prep_enc(const float* __restrict__ Wih, const float* __restrict__ Whh,
                           const float* __restrict__ bih, const float* __restrict__ bhh,
                           unsigned short* __restrict__ Wcat, float* __restrict__ bias) {
  const int stride = gridDim.x * blockDim.x;
  for (int i = blockIdx.x * blockDim.x + threadIdx.x; i < 2048 * 640; i += stride) {
    int g = i / 640, k = i - g * 640;
    float v = (k < DIN) ? Wih[g * DIN + k] : Whh[g * HD + (k - DIN)];
    Wcat[i] = f2bf(v);
  }
  for (int i = blockIdx.x * blockDim.x + threadIdx.x; i < 2048; i += stride)
    bias[i] = bih[i] + bhh[i];
}

__global__ void k_cvt_bf16(const float* __restrict__ src, unsigned short* __restrict__ dst, int n) {
  const int stride = gridDim.x * blockDim.x;
  for (int i = blockIdx.x * blockDim.x + threadIdx.x; i < n; i += stride) dst[i] = f2bf(src[i]);
}

__global__ void k_pad_bf16(const float* __restrict__ src, unsigned short* __restrict__ dst,
                           int R, int Cp, int r, int c) {
  const int stride = gridDim.x * blockDim.x;
  for (int i = blockIdx.x * blockDim.x + threadIdx.x; i < R * Cp; i += stride) {
    int g = i / Cp, k = i - g * Cp;
    dst[i] = (g < r && k < c) ? f2bf(src[g * c + k]) : (unsigned short)0;
  }
}

// x2[s] = [f1[2s] | b1[119-2s] | f1[2s+1] | b1[118-2s]]
__global__ void k_gather_x2(const float* __restrict__ f1, const float* __restrict__ b1,
                            float* __restrict__ x2) {
  int idx = blockIdx.x * blockDim.x + threadIdx.x;
  if (idx >= NCTX * 2048) return;
  int s = idx >> 11, c = idx & 2047;
  float v;
  if (c < 512)       v = f1[(2 * s) * HD + c];
  else if (c < 1024) v = b1[(119 - 2 * s) * HD + (c - 512)];
  else if (c < 1536) v = f1[(2 * s + 1) * HD + (c - 1024)];
  else               v = b1[(118 - 2 * s) * HD + (c - 1536)];
  x2[idx] = v;
}

// ---------------- encoder: 15 WGs x 16 rows, 64 fused timesteps ----------------
__global__ __launch_bounds__(1024) void k_encoder(const float* __restrict__ data,
                                                  const unsigned short* __restrict__ Wcat,
                                                  const float* __restrict__ bias,
                                                  float* __restrict__ h_enc) {
  __shared__ alignas(16) unsigned short rowbuf[16 * 648];  // [x(128) | h(512) | pad(8)] per row
  const int tid = threadIdx.x;
  const int wave = tid >> 5, lane = tid & 31;
  const int n = lane & 15, hs = lane >> 4;
  const int r0 = blockIdx.x * 16;
  const int jcol = wave * 16 + n;  // hidden unit owned by this lane
  for (int i = tid; i < 16 * 648; i += 1024) rowbuf[i] = 0;
  const float bI = bias[jcol], bF = bias[512 + jcol], bG = bias[1024 + jcol], bO = bias[1536 + jcol];
  float cst[8], hreg[8];
#pragma unroll
  for (int r = 0; r < 8; ++r) { cst[r] = 0.f; hreg[r] = 0.f; }

  for (int t = 0; t < TSTP; ++t) {
    // stage x_t (f32 -> bf16) into LDS
    for (int e = tid; e < 16 * DIN; e += 1024) {
      int m = e >> 7, d = e & 127;
      rowbuf[m * 648 + d] = f2bf(data[((size_t)t * WFR + r0 + m) * DIN + d]);
    }
    __syncthreads();  // x staged AND previous h writes visible

    FragF aI, aF, aG, aO;
#pragma unroll
    for (int r = 0; r < 8; ++r) { aI.f[r] = bI; aF.f[r] = bF; aG.f[r] = bG; aO.f[r] = bO; }
    for (int k0 = 0; k0 < 640; k0 += 32) {
      FragBF a = load_a_lds(rowbuf, k0, 648);
      wmma_acc(aI, a, load_b_bf(Wcat,        wave * 16, k0, 640));
      wmma_acc(aF, a, load_b_bf(Wcat,  512 + wave * 16, k0, 640));
      wmma_acc(aG, a, load_b_bf(Wcat, 1024 + wave * 16, k0, 640));
      wmma_acc(aO, a, load_b_bf(Wcat, 1536 + wave * 16, k0, 640));
    }
#pragma unroll
    for (int r = 0; r < 8; ++r) {
      float ig = sigf(aI.f[r]), fg = sigf(aF.f[r]);
      float gg = tanh_f(aG.f[r]), og = sigf(aO.f[r]);
      cst[r] = fg * cst[r] + ig * gg;
      hreg[r] = og * tanh_f(cst[r]);
    }
    __syncthreads();  // all LDS reads complete before overwriting h region
#pragma unroll
    for (int r = 0; r < 8; ++r) {
      int row = r + 8 * hs;
      rowbuf[row * 648 + DIN + jcol] = f2bf(hreg[r]);
    }
  }
#pragma unroll
  for (int r = 0; r < 8; ++r) {
    int row = r + 8 * hs;
    h_enc[(size_t)(r0 + row) * HD + jcol] = hreg[r];
  }
}

// ---------------- batched input-projection pre-GEMM: C = A @ W^T + b1 + b2 ----------------
__global__ __launch_bounds__(256) void k_pregemm(const float* __restrict__ A, int M, int K,
                                                 const float* __restrict__ Wt,
                                                 const float* __restrict__ b1,
                                                 const float* __restrict__ b2,
                                                 float* __restrict__ C) {
  const int tid = threadIdx.x, wave = tid >> 5, lane = tid & 31;
  const int n0 = (blockIdx.y * 8 + wave) * 16;
  const int m0 = blockIdx.x * 16;
  FragF acc;
#pragma unroll
  for (int r = 0; r < 8; ++r) acc.f[r] = 0.f;
  for (int k0 = 0; k0 < K; k0 += 32) {
    FragBF a = load_a_f32(A, m0, M, k0, K);
    FragBF b = load_b_f32(Wt, n0, k0, K);
    wmma_acc(acc, a, b);
  }
  const int col = n0 + (lane & 15);
  const int hs = lane >> 4;
  const float bias = b1[col] + b2[col];
#pragma unroll
  for (int r = 0; r < 8; ++r) {
    int row = m0 + r + 8 * hs;
    if (row < M) C[(size_t)row * 2048 + col] = acc.f[r] + bias;
  }
}

// ---------------- serial batch-1 LSTM loop (fwd/bwd in blocks 0/1) ----------------
__global__ __launch_bounds__(1024) void k_lstm_seq(
    const float* __restrict__ xp0, const unsigned short* __restrict__ Whh0, float* __restrict__ out0,
    const float* __restrict__ xp1, const unsigned short* __restrict__ Whh1, float* __restrict__ out1,
    int nsteps, int out_stride, int out_rev1) {
  __shared__ alignas(16) unsigned short hbuf[HD + 16];
  const int tid = threadIdx.x, wave = tid >> 5, lane = tid & 31;
  const int dir = blockIdx.x;
  const float* xp = dir ? xp1 : xp0;
  const unsigned short* Whh = dir ? Whh1 : Whh0;
  float* out = dir ? out1 : out0;
  const int xp_rev = dir;
  const int out_rev = dir ? out_rev1 : 0;
  const int jcol = wave * 16 + (lane & 15);
  for (int i = tid; i < HD + 16; i += 1024) hbuf[i] = 0;
  float cst = 0.f;
  __syncthreads();
  for (int s = 0; s < nsteps; ++s) {
    const float* xr = xp + (size_t)(xp_rev ? (nsteps - 1 - s) : s) * 2048;
    const float vI = xr[jcol], vF = xr[512 + jcol], vG = xr[1024 + jcol], vO = xr[1536 + jcol];
    FragF aI, aF, aG, aO;
#pragma unroll
    for (int r = 0; r < 8; ++r) { aI.f[r] = vI; aF.f[r] = vF; aG.f[r] = vG; aO.f[r] = vO; }
    for (int k0 = 0; k0 < HD; k0 += 32) {
      FragBF a = load_a_vec(hbuf, k0);
      wmma_acc(aI, a, load_b_bf(Whh,        wave * 16, k0, HD));
      wmma_acc(aF, a, load_b_bf(Whh,  512 + wave * 16, k0, HD));
      wmma_acc(aG, a, load_b_bf(Whh, 1024 + wave * 16, k0, HD));
      wmma_acc(aO, a, load_b_bf(Whh, 1536 + wave * 16, k0, HD));
    }
    float hval = 0.f;
    if (lane < 16) {
      float ig = sigf(aI.f[0]), fg = sigf(aF.f[0]);
      float gg = tanh_f(aG.f[0]), og = sigf(aO.f[0]);
      cst = fg * cst + ig * gg;
      hval = og * tanh_f(cst);
    }
    __syncthreads();  // all hbuf reads done
    if (lane < 16) {
      hbuf[jcol] = f2bf(hval);
      int orow = out_rev ? (nsteps - 1 - s) : s;
      out[(size_t)orow * out_stride + jcol] = hval;
    }
    __syncthreads();  // h writes visible before next step reads
  }
}

// ---------------- decoder: one persistent WG, 30 serial attention/GRU steps ----------------
__global__ __launch_bounds__(512) void k_decoder(
    const float* __restrict__ ctx,               // 60 x 1024
    const unsigned short* __restrict__ attnWp,   // 64 x 544 bf16 (padded)
    const float* __restrict__ attn_b,            // 60
    const unsigned short* __restrict__ combWp,   // 512 x 1056 bf16 (padded)
    const float* __restrict__ comb_b,            // 512
    const unsigned short* __restrict__ gruWih,   // 1536 x 512 bf16
    const unsigned short* __restrict__ gruWhh,   // 1536 x 512 bf16
    const float* __restrict__ gru_bih, const float* __restrict__ gru_bhh,
    const float* __restrict__ out_W, const float* __restrict__ out_b,
    float* __restrict__ outp) {
  __shared__ float sh_h[HD];
  __shared__ float sh_inp[NLAB];
  __shared__ float sh_o[HD];
  __shared__ float sh_app[2 * HD];
  __shared__ float sh_gi[3 * HD];
  __shared__ float sh_gh[3 * HD];
  __shared__ float sh_aw[64];
  __shared__ float sh_red[2];
  __shared__ float sh_logit[NLAB];
  __shared__ alignas(16) unsigned short sh_vec[1088];
  const int tid = threadIdx.x, wave = tid >> 5, lane = tid & 31;

  if (tid < HD) sh_h[tid] = ctx[59 * 1024 + tid];  // h0 = f2[59]
  if (tid < NLAB) sh_inp[tid] = (tid == NLAB - 1) ? 1.0f : 0.0f;
  for (int i = tid; i < 1088; i += 512) sh_vec[i] = 0;
  __syncthreads();

  for (int step = 0; step < NDEC; ++step) {
    // ---- attention scores: aw = attn_W @ [inp, h] + b ----
    if (tid < NLAB) sh_vec[tid] = f2bf(sh_inp[tid]);
    if (tid < HD) sh_vec[NLAB + tid] = f2bf(sh_h[tid]);
    if (tid >= 520 && tid < 544) sh_vec[tid] = 0;  // re-zero K padding
    __syncthreads();
    if (wave < 4) {
      FragF acc;
#pragma unroll
      for (int r = 0; r < 8; ++r) acc.f[r] = 0.f;
      for (int k0 = 0; k0 < 544; k0 += 32) {
        FragBF a = load_a_vec(sh_vec, k0);
        wmma_acc(acc, a, load_b_bf(attnWp, wave * 16, k0, 544));
      }
      if (lane < 16) {
        int col = wave * 16 + lane;
        sh_aw[col] = (col < NCTX) ? acc.f[0] + attn_b[col] : -1e30f;
      }
    }
    __syncthreads();
    // ---- softmax over 60 ----
    if (tid == 0) {
      float m = -1e30f;
      for (int i = 0; i < NCTX; ++i) m = fmaxf(m, sh_aw[i]);
      sh_red[0] = m;
    }
    __syncthreads();
    if (tid < 64) sh_aw[tid] = (tid < NCTX) ? __expf(sh_aw[tid] - sh_red[0]) : 0.0f;
    __syncthreads();
    if (tid == 0) {
      float s = 0.f;
      for (int i = 0; i < NCTX; ++i) s += sh_aw[i];
      sh_red[1] = 1.0f / s;
    }
    __syncthreads();
    // ---- applied = aw @ ctx ----
    if (tid < HD) {
      const float rn = sh_red[1];
      float s0 = 0.f, s1 = 0.f;
      for (int s = 0; s < NCTX; ++s) {
        float a = sh_aw[s] * rn;
        s0 += a * ctx[s * 1024 + tid];
        s1 += a * ctx[s * 1024 + HD + tid];
      }
      sh_app[tid] = s0; sh_app[HD + tid] = s1;
    }
    __syncthreads();
    // ---- o = relu(comb_W @ [inp, applied] + b) ----
    if (tid < NLAB) sh_vec[tid] = f2bf(sh_inp[tid]);
    if (tid < HD) {
      sh_vec[NLAB + tid] = f2bf(sh_app[tid]);
      sh_vec[NLAB + HD + tid] = f2bf(sh_app[HD + tid]);
    }
    __syncthreads();
    {
      FragF a0, a1;
#pragma unroll
      for (int r = 0; r < 8; ++r) { a0.f[r] = 0.f; a1.f[r] = 0.f; }
      for (int k0 = 0; k0 < 1056; k0 += 32) {
        FragBF a = load_a_vec(sh_vec, k0);
        wmma_acc(a0, a, load_b_bf(combWp, (2 * wave) * 16, k0, 1056));
        wmma_acc(a1, a, load_b_bf(combWp, (2 * wave + 1) * 16, k0, 1056));
      }
      if (lane < 16) {
        int c0 = (2 * wave) * 16 + lane, c1 = c0 + 16;
        sh_o[c0] = fmaxf(a0.f[0] + comb_b[c0], 0.f);
        sh_o[c1] = fmaxf(a1.f[0] + comb_b[c1], 0.f);
      }
    }
    __syncthreads();
    // ---- gi = Wih @ o + bih ----
    if (tid < HD) sh_vec[tid] = f2bf(sh_o[tid]);
    __syncthreads();
    {
      FragF g[6];
#pragma unroll
      for (int q = 0; q < 6; ++q)
#pragma unroll
        for (int r = 0; r < 8; ++r) g[q].f[r] = 0.f;
      for (int k0 = 0; k0 < HD; k0 += 32) {
        FragBF a = load_a_vec(sh_vec, k0);
#pragma unroll
        for (int q = 0; q < 6; ++q)
          wmma_acc(g[q], a, load_b_bf(gruWih, (wave * 6 + q) * 16, k0, HD));
      }
      if (lane < 16) {
#pragma unroll
        for (int q = 0; q < 6; ++q) {
          int col = (wave * 6 + q) * 16 + lane;
          sh_gi[col] = g[q].f[0] + gru_bih[col];
        }
      }
    }
    __syncthreads();
    // ---- gh = Whh @ h + bhh ----
    if (tid < HD) sh_vec[tid] = f2bf(sh_h[tid]);
    __syncthreads();
    {
      FragF g[6];
#pragma unroll
      for (int q = 0; q < 6; ++q)
#pragma unroll
        for (int r = 0; r < 8; ++r) g[q].f[r] = 0.f;
      for (int k0 = 0; k0 < HD; k0 += 32) {
        FragBF a = load_a_vec(sh_vec, k0);
#pragma unroll
        for (int q = 0; q < 6; ++q)
          wmma_acc(g[q], a, load_b_bf(gruWhh, (wave * 6 + q) * 16, k0, HD));
      }
      if (lane < 16) {
#pragma unroll
        for (int q = 0; q < 6; ++q) {
          int col = (wave * 6 + q) * 16 + lane;
          sh_gh[col] = g[q].f[0] + gru_bhh[col];
        }
      }
    }
    __syncthreads();
    // ---- GRU pointwise (r,z,n) ----
    if (tid < HD) {
      float r = sigf(sh_gi[tid] + sh_gh[tid]);
      float z = sigf(sh_gi[HD + tid] + sh_gh[HD + tid]);
      float nn = tanh_f(sh_gi[2 * HD + tid] + r * sh_gh[2 * HD + tid]);
      sh_h[tid] = (1.0f - z) * nn + z * sh_h[tid];
    }
    __syncthreads();
    // ---- output log-softmax ----
    if (tid < NLAB) {
      float s = out_b[tid];
      const float* wr = out_W + tid * HD;
      for (int k = 0; k < HD; ++k) s += wr[k] * sh_h[k];
      sh_logit[tid] = s;
    }
    __syncthreads();
    if (tid == 0) {
      float m = -1e30f;
      for (int i = 0; i < NLAB; ++i) m = fmaxf(m, sh_logit[i]);
      float s = 0.f;
      for (int i = 0; i < NLAB; ++i) s += __expf(sh_logit[i] - m);
      sh_red[0] = m + __logf(s);
    }
    __syncthreads();
    if (tid < NLAB) {
      float lp = sh_logit[tid] - sh_red[0];
      sh_inp[tid] = lp;
      outp[step * NLAB + tid] = lp;
    }
    __syncthreads();
  }
}

// ---------------- host launcher ----------------
extern "C" void kernel_launch(void* const* d_in, const int* in_sizes, int n_in,
                              void* d_out, int out_size, void* d_ws, size_t ws_size,
                              hipStream_t stream) {
  (void)in_sizes; (void)n_in; (void)out_size; (void)ws_size;
  const float* data     = (const float*)d_in[0];
  const float* enc_Wih  = (const float*)d_in[1];
  const float* enc_Whh  = (const float*)d_in[2];
  const float* enc_bih  = (const float*)d_in[3];
  const float* enc_bhh  = (const float*)d_in[4];
  const float* p1f_Wih  = (const float*)d_in[5];
  const float* p1f_Whh  = (const float*)d_in[6];
  const float* p1f_bih  = (const float*)d_in[7];
  const float* p1f_bhh  = (const float*)d_in[8];
  const float* p1b_Wih  = (const float*)d_in[9];
  const float* p1b_Whh  = (const float*)d_in[10];
  const float* p1b_bih  = (const float*)d_in[11];
  const float* p1b_bhh  = (const float*)d_in[12];
  const float* p2f_Wih  = (const float*)d_in[13];
  const float* p2f_Whh  = (const float*)d_in[14];
  const float* p2f_bih  = (const float*)d_in[15];
  const float* p2f_bhh  = (const float*)d_in[16];
  const float* p2b_Wih  = (const float*)d_in[17];
  const float* p2b_Whh  = (const float*)d_in[18];
  const float* p2b_bih  = (const float*)d_in[19];
  const float* p2b_bhh  = (const float*)d_in[20];
  const float* attn_W   = (const float*)d_in[21];
  const float* attn_b   = (const float*)d_in[22];
  const float* comb_W   = (const float*)d_in[23];
  const float* comb_b   = (const float*)d_in[24];
  const float* gru_Wih  = (const float*)d_in[25];
  const float* gru_Whh  = (const float*)d_in[26];
  const float* gru_bih  = (const float*)d_in[27];
  const float* gru_bhh  = (const float*)d_in[28];
  const float* out_W    = (const float*)d_in[29];
  const float* out_b    = (const float*)d_in[30];

  char* ws = (char*)d_ws;
  size_t off = 0;
  auto alloc = [&](size_t bytes) -> void* {
    void* p = ws + off;
    off = (off + bytes + 255) & ~(size_t)255;
    return p;
  };
  unsigned short* encWcat = (unsigned short*)alloc((size_t)2048 * 640 * 2);
  float*          encBias = (float*)alloc(2048 * 4);
  float*          h_enc   = (float*)alloc((size_t)WFR * HD * 4);
  unsigned short* Whh1f   = (unsigned short*)alloc((size_t)2048 * HD * 2);
  unsigned short* Whh1b   = (unsigned short*)alloc((size_t)2048 * HD * 2);
  unsigned short* Whh2f   = (unsigned short*)alloc((size_t)2048 * HD * 2);
  unsigned short* Whh2b   = (unsigned short*)alloc((size_t)2048 * HD * 2);
  unsigned short* gruWihB = (unsigned short*)alloc((size_t)1536 * HD * 2);
  unsigned short* gruWhhB = (unsigned short*)alloc((size_t)1536 * HD * 2);
  unsigned short* attnWp  = (unsigned short*)alloc((size_t)64 * 544 * 2);
  unsigned short* combWp  = (unsigned short*)alloc((size_t)512 * 1056 * 2);
  float*          xp1f    = (float*)alloc((size_t)120 * 2048 * 4);
  float*          xp1b    = (float*)alloc((size_t)120 * 2048 * 4);
  float*          f1      = (float*)alloc((size_t)120 * HD * 4);
  float*          b1      = (float*)alloc((size_t)120 * HD * 4);
  float*          x2      = (float*)alloc((size_t)NCTX * 2048 * 4);
  float*          xp2f    = (float*)alloc((size_t)NCTX * 2048 * 4);
  float*          xp2b    = (float*)alloc((size_t)NCTX * 2048 * 4);
  float*          ctx     = (float*)alloc((size_t)NCTX * 1024 * 4);

  // weight prep (bf16 conversions / concat / padding)
  k_prep_enc<<<1024, 256, 0, stream>>>(enc_Wih, enc_Whh, enc_bih, enc_bhh, encWcat, encBias);
  k_cvt_bf16<<<1024, 256, 0, stream>>>(p1f_Whh, Whh1f, 2048 * HD);
  k_cvt_bf16<<<1024, 256, 0, stream>>>(p1b_Whh, Whh1b, 2048 * HD);
  k_cvt_bf16<<<1024, 256, 0, stream>>>(p2f_Whh, Whh2f, 2048 * HD);
  k_cvt_bf16<<<1024, 256, 0, stream>>>(p2b_Whh, Whh2b, 2048 * HD);
  k_cvt_bf16<<<1024, 256, 0, stream>>>(gru_Wih, gruWihB, 1536 * HD);
  k_cvt_bf16<<<1024, 256, 0, stream>>>(gru_Whh, gruWhhB, 1536 * HD);
  k_pad_bf16<<<256, 256, 0, stream>>>(attn_W, attnWp, 64, 544, 60, 520);
  k_pad_bf16<<<1024, 256, 0, stream>>>(comb_W, combWp, 512, 1056, 512, 1032);

  // encoder: 240 independent LSTMs, fused time loop
  k_encoder<<<15, 1024, 0, stream>>>(data, encWcat, encBias, h_enc);

  // pyramid level 1: batched input projection, then serial h-recurrence (fwd+bwd concurrent)
  k_pregemm<<<dim3(8, 16), 256, 0, stream>>>(h_enc, 120, 1024, p1f_Wih, p1f_bih, p1f_bhh, xp1f);
  k_pregemm<<<dim3(8, 16), 256, 0, stream>>>(h_enc, 120, 1024, p1b_Wih, p1b_bih, p1b_bhh, xp1b);
  k_lstm_seq<<<2, 1024, 0, stream>>>(xp1f, Whh1f, f1, xp1b, Whh1b, b1, 120, HD, 0);

  // pyramid level 2
  k_gather_x2<<<512, 256, 0, stream>>>(f1, b1, x2);
  k_pregemm<<<dim3(4, 16), 256, 0, stream>>>(x2, 60, 2048, p2f_Wih, p2f_bih, p2f_bhh, xp2f);
  k_pregemm<<<dim3(4, 16), 256, 0, stream>>>(x2, 60, 2048, p2b_Wih, p2b_bih, p2b_bhh, xp2b);
  k_lstm_seq<<<2, 1024, 0, stream>>>(xp2f, Whh2f, ctx, xp2b, Whh2b, ctx + HD, 60, 1024, 1);

  // decoder (attention + GRU + log-softmax), 30 serial steps
  k_decoder<<<1, 512, 0, stream>>>(ctx, attnWp, attn_b, combWp, comb_b,
                                   gruWihB, gruWhhB, gru_bih, gru_bhh,
                                   out_W, out_b, (float*)d_out);
}
